// NTU_4243427689282
// MI455X (gfx1250) — matrix operand order repaired and checked by reading, not compiled, not run on hardware
//
#include <hip/hip_runtime.h>
#include <stdint.h>

// ---------------------------------------------------------------------------
// Types for CDNA5 WMMA (wave32): bf16 A/B fragments, f32 accumulator.
// ---------------------------------------------------------------------------
typedef __attribute__((ext_vector_type(8)))  float  v8f;
typedef __attribute__((ext_vector_type(16))) __bf16 v16bf;
typedef __attribute__((ext_vector_type(8)))  __bf16 v8bf;

#define KNN 9
#define EPSV 1e-5f

__device__ __forceinline__ unsigned short f32_to_bf16_rne(float f) {
  union { float f; unsigned u; } c; c.f = f;
  unsigned u = c.u;
  unsigned r = u + 0x7fffu + ((u >> 16) & 1u);   // round-to-nearest-even
  return (unsigned short)(r >> 16);
}

// ---------------------------------------------------------------------------
// Farthest point sampling: one block per (frame, batch). Serial m-step scan,
// parallel distance update + LDS argmax tree per step.
// idx[0] = 0; idx[i] = argmax d2 after incorporating idx[i-1]   (matches scan)
// ---------------------------------------------------------------------------
__global__ void fps_kernel(const float* __restrict__ xyz, int Lsrc, int n, int m,
                           int rad, int ts, int* __restrict__ fidx, int lout) {
  int f = blockIdx.x, bb = blockIdx.y;
  const float* P = xyz + (((size_t)bb * Lsrc + (rad + f * ts)) * (size_t)n) * 3;
  int* out = fidx + ((size_t)bb * lout + f) * m;
  __shared__ float d2[2048];
  __shared__ float rv[256];
  __shared__ int   ri[256];
  int tid = threadIdx.x;
  for (int i = tid; i < n; i += 256) d2[i] = 1e10f;
  __syncthreads();
  int last = 0;
  for (int it = 0; it < m; ++it) {
    if (tid == 0) out[it] = last;
    float lx = P[3 * last], ly = P[3 * last + 1], lz = P[3 * last + 2];
    float best = -1.0f; int bi = 0;
    for (int i = tid; i < n; i += 256) {
      float dx = P[3 * i] - lx, dy = P[3 * i + 1] - ly, dz = P[3 * i + 2] - lz;
      float dd = dx * dx + dy * dy + dz * dz;
      float cur = fminf(d2[i], dd);
      d2[i] = cur;
      if (cur > best) { best = cur; bi = i; }
    }
    rv[tid] = best; ri[tid] = bi;
    __syncthreads();
    for (int s = 128; s > 0; s >>= 1) {
      if (tid < s) {
        if (rv[tid + s] > rv[tid] ||
            (rv[tid + s] == rv[tid] && ri[tid + s] < ri[tid])) {
          rv[tid] = rv[tid + s]; ri[tid] = ri[tid + s];
        }
      }
      __syncthreads();
    }
    last = ri[0];
    __syncthreads();
  }
}

// anchor[b][f][i][:] = xyz[b][rad + f*ts][fidx[b][f][i]][:]
__global__ void gather_anchor_k(const float* __restrict__ xyz, const int* __restrict__ fidx,
                                float* __restrict__ anchor, int Lsrc, int n, int m,
                                int rad, int ts, int lout) {
  int t = blockIdx.x * blockDim.x + threadIdx.x;
  int total = 2 * lout * m;
  if (t >= total) return;
  int pm = t % m; int f = (t / m) % lout; int bb = t / (m * lout);
  int pi = fidx[t];
  const float* p = xyz + (((size_t)bb * Lsrc + (rad + f * ts)) * (size_t)n + pi) * 3;
  float* o = anchor + (size_t)t * 3;
  o[0] = p[0]; o[1] = p[1]; o[2] = p[2];
}

// Ball query: first K indices (ascending) with d2 < r^2; pad with first / 0.
__global__ void ball_query_k(const float* __restrict__ anchor, const float* __restrict__ xyz,
                             int* __restrict__ idx, int lout, int tk, int m, int n,
                             int Lsrc, int ts, float r2) {
  int pm = blockIdx.x * blockDim.x + threadIdx.x;
  if (pm >= m) return;
  int w = blockIdx.y;
  int bf = blockIdx.z; int f = bf % lout; int bb = bf / lout;
  const float* a = anchor + (((size_t)bb * lout + f) * m + pm) * 3;
  float ax = a[0], ay = a[1], az = a[2];
  const float* P = xyz + ((size_t)bb * Lsrc + (f * ts + w)) * (size_t)n * 3;
  int out[KNN]; int cnt = 0;
  for (int j = 0; j < n && cnt < KNN; ++j) {
    float dx = P[3 * j] - ax, dy = P[3 * j + 1] - ay, dz = P[3 * j + 2] - az;
    if (dx * dx + dy * dy + dz * dz < r2) out[cnt++] = j;
  }
  int pad = (cnt > 0) ? out[0] : 0;
  for (int k = cnt; k < KNN; ++k) out[k] = pad;
  int* o = idx + ((((size_t)bb * lout + f) * tk + w) * m + pm) * (size_t)KNN;
  for (int k = 0; k < KNN; ++k) o[k] = out[k];
}

// ---------------------------------------------------------------------------
// Stage-1 B pack: B1t[col][k] (bf16, zero-padded to Kpad), col=(b,f,w,pm).
// k 0..2 : sum_k grouped_xyz - 9*anchor ; k 3.. : sum_k gathered feats.
// (K-pooling folded before the GEMM: 9x FLOP reduction.)
// ---------------------------------------------------------------------------
__global__ void pack_b1_k(const float* __restrict__ xyz, const float* __restrict__ feats,
                          const float* __restrict__ anchor, const int* __restrict__ idx,
                          unsigned short* __restrict__ B1t, int lout, int tk, int m, int n,
                          int cin, int Kpad, int Lsrc, int ts) {
  int col = blockIdx.x;
  int pm = col % m; int w = (col / m) % tk;
  int f = (col / (m * tk)) % lout; int bb = col / (m * tk * lout);
  __shared__ int   sidx[KNN];
  __shared__ float sanch[3];
  int tid = threadIdx.x;
  if (tid < KNN) sidx[tid] = idx[(size_t)col * KNN + tid];
  if (tid >= KNN && tid < KNN + 3)
    sanch[tid - KNN] = anchor[(((size_t)bb * lout + f) * m + pm) * 3 + (tid - KNN)];
  __syncthreads();
  const float* P = xyz + ((size_t)bb * Lsrc + (f * ts + w)) * (size_t)n * 3;
  const float* F = feats ? feats + ((size_t)bb * Lsrc + (f * ts + w)) * (size_t)n * cin : nullptr;
  unsigned short* orow = B1t + (size_t)col * Kpad;
  int C = 3 + cin;
  for (int ch = tid; ch < Kpad; ch += blockDim.x) {
    float s = 0.0f;
    if (ch < 3) {
      #pragma unroll
      for (int k = 0; k < KNN; ++k) s += P[3 * sidx[k] + ch];
      s -= (float)KNN * sanch[ch];
    } else if (ch < C) {
      int cc = ch - 3;
      #pragma unroll
      for (int k = 0; k < KNN; ++k) s += F[(size_t)sidx[k] * cin + cc];
    }
    orow[ch] = f32_to_bf16_rne(s);
  }
}

// A1[row][k] = [Wd | Wf], bf16, zero-padded to Mpad x Kpad
__global__ void pack_a1_k(const float* __restrict__ Wd, const float* __restrict__ Wf,
                          unsigned short* __restrict__ A, int mid, int cin, int Kpad, int Mpad) {
  int t = blockIdx.x * blockDim.x + threadIdx.x;
  int total = Mpad * Kpad;
  if (t >= total) return;
  int k = t % Kpad, row = t / Kpad;
  float v = 0.0f;
  if (row < mid) {
    if (k < 3) v = Wd[row * 3 + k];
    else if (k - 3 < cin) v = Wf[(size_t)row * cin + (k - 3)];
  }
  A[t] = f32_to_bf16_rne(v);
}

// A2[row][k] = Wt, bf16, zero-padded
__global__ void pack_a2_k(const float* __restrict__ Wt, unsigned short* __restrict__ A,
                          int cout, int Kdim, int Kpad, int Mpad) {
  int t = blockIdx.x * blockDim.x + threadIdx.x;
  int total = Mpad * Kpad;
  if (t >= total) return;
  int k = t % Kpad, row = t / Kpad;
  float v = (row < cout && k < Kdim) ? Wt[(size_t)row * Kdim + k] : 0.0f;
  A[t] = f32_to_bf16_rne(v);
}

// Stage-2 B pack: BN + ReLU + (t,m)->(m) repack, bf16, zero-padded.
// col2=(b,f,pm); channel c = t*mid + o reads out1[((b,f,t),pm)][o].
__global__ void pack_b2_k(const float* __restrict__ out1, const float* __restrict__ g,
                          const float* __restrict__ be, const float* __restrict__ rm,
                          const float* __restrict__ rv, unsigned short* __restrict__ B2t,
                          int lout, int tk, int m, int mid, int Kpad) {
  int col2 = blockIdx.x;
  int pm = col2 % m; int lf = col2 / m;  // lf = bb*lout + f
  int C = tk * mid;
  unsigned short* orow = B2t + (size_t)col2 * Kpad;
  for (int c = threadIdx.x; c < Kpad; c += blockDim.x) {
    float v = 0.0f;
    if (c < C) {
      int t = c / mid, o = c % mid;
      float x = out1[(((size_t)lf * tk + t) * m + pm) * mid + o];
      float inv = g[c] * rsqrtf(rv[c] + EPSV);
      v = fmaxf(x * inv + (be[c] - rm[c] * inv), 0.0f);
    }
    orow[c] = f32_to_bf16_rne(v);
  }
}

// ---------------------------------------------------------------------------
// WMMA GEMM: C[N][M] (f32) = A[Mpad x Kpad] * Bt[N x Kpad]^T, bf16 inputs.
// Block = 128 threads = 4 wave32s; each wave owns a 16x64 strip of C
// (4 N-tiles) so one A fragment feeds 4 WMMAs per K-step (4x FLOP/A-byte).
// N is guaranteed to be a multiple of 256 by construction (m >= 128, B=2),
// so no tail guards are needed and EXEC stays all-ones for every WMMA.
// Fragment layouts used here (contiguous k, same in A and B; not the ISA table order):
//   A lane L (M=L%16): k = (L/16)*16 + j                  -> two 16B loads
//   B lane L (N=L%16): k = (L/16)*16 + j                  -> one 32B load
//   D VGPR v: M = v + (L>=16 ? 8 : 0), N = L%16
// ---------------------------------------------------------------------------
__global__ void gemm_wmma_bf16(const unsigned short* __restrict__ Ap,
                               const unsigned short* __restrict__ Btp,
                               float* __restrict__ C, const float* __restrict__ bias,
                               int M, int N, int Kpad, int relu) {
  const __bf16* A  = (const __bf16*)Ap;
  const __bf16* Bt = (const __bf16*)Btp;
  int lane = threadIdx.x & 31;
  int wave = threadIdx.x >> 5;
  int n0 = blockIdx.y * 256 + wave * 64;   // 4 waves x 64 cols = 256 cols/block
  int mtile = blockIdx.x;
  v8f acc[4];
  #pragma unroll
  for (int t = 0; t < 4; ++t) acc[t] = (v8f){0.f,0.f,0.f,0.f,0.f,0.f,0.f,0.f};
  int hi8 = (lane >> 4) << 3;
  const size_t bstride = (size_t)16 * Kpad;  // one N-tile of Bt
  const __bf16* Arow = A  + (size_t)(mtile * 16 + (lane & 15)) * Kpad;
  const __bf16* B0   = Bt + (size_t)(n0 + (lane & 15)) * Kpad + ((lane >> 4) << 4);
  for (int k = 0; k < Kpad; k += 32) {
    __builtin_prefetch(Arow + k + 256, 0, 3);   // global_prefetch_b8, near cache
    __builtin_prefetch(B0   + k + 256, 0, 3);
    __builtin_prefetch(B0 + bstride     + k + 256, 0, 3);
    __builtin_prefetch(B0 + 2 * bstride + k + 256, 0, 3);
    __builtin_prefetch(B0 + 3 * bstride + k + 256, 0, 3);
    v8bf a0 = *(const v8bf*)(Arow + k + 2 * hi8);
    v8bf a1 = *(const v8bf*)(Arow + k + 8 + 2 * hi8);
    v16bf a;
    #pragma unroll
    for (int j = 0; j < 8; ++j) { a[j] = a0[j]; a[j + 8] = a1[j]; }
    v16bf b0 = *(const v16bf*)(B0 + k);
    v16bf b1 = *(const v16bf*)(B0 + bstride + k);
    v16bf b2 = *(const v16bf*)(B0 + 2 * bstride + k);
    v16bf b3 = *(const v16bf*)(B0 + 3 * bstride + k);
    acc[0] = __builtin_amdgcn_wmma_f32_16x16x32_bf16(false, a, false, b0,
                                                     (short)0, acc[0], false, false);
    acc[1] = __builtin_amdgcn_wmma_f32_16x16x32_bf16(false, a, false, b1,
                                                     (short)0, acc[1], false, false);
    acc[2] = __builtin_amdgcn_wmma_f32_16x16x32_bf16(false, a, false, b2,
                                                     (short)0, acc[2], false, false);
    acc[3] = __builtin_amdgcn_wmma_f32_16x16x32_bf16(false, a, false, b3,
                                                     (short)0, acc[3], false, false);
  }
  int mbase = mtile * 16 + ((lane >> 4) << 3);
  #pragma unroll
  for (int t = 0; t < 4; ++t) {
    int n = n0 + 16 * t + (lane & 15);
    #pragma unroll
    for (int v = 0; v < 8; ++v) {
      int mr = mbase + v;
      if (mr < M) {
        float val = acc[t][v] + (bias ? bias[mr] : 0.0f);
        if (relu) val = fmaxf(val, 0.0f);
        C[(size_t)n * M + mr] = val;
      }
    }
  }
}

// Head: f[b][ch] = max_l mean_m feats[b][l][m][ch]
__global__ void head_reduce_k(const float* __restrict__ feats, float* __restrict__ f,
                              int lfin, int mfin, int cfin) {
  int t = blockIdx.x * blockDim.x + threadIdx.x;
  int total = 2 * cfin;
  if (t >= total) return;
  int ch = t % cfin; int bb = t / cfin;
  float best = -3.4e38f;
  for (int l = 0; l < lfin; ++l) {
    float s = 0.0f;
    const float* base = feats + (((size_t)bb * lfin + l) * mfin) * (size_t)cfin + ch;
    for (int i = 0; i < mfin; ++i) s += base[(size_t)i * cfin];
    best = fmaxf(best, s / (float)mfin);
  }
  f[t] = best;
}

__global__ void head_fc_k(const float* __restrict__ f, const float* __restrict__ W,
                          const float* __restrict__ b, float* __restrict__ out,
                          int nclass, int cfin) {
  int t = blockIdx.x * blockDim.x + threadIdx.x;
  if (t >= 2 * nclass) return;
  int o = t % nclass; int bb = t / nclass;
  float s = b[o];
  const float* fr = f + (size_t)bb * cfin;
  const float* wr = W + (size_t)o * cfin;
  for (int c = 0; c < cfin; ++c) s += fr[c] * wr[c];
  out[t] = s;
}

// ---------------------------------------------------------------------------
// Host orchestration
// ---------------------------------------------------------------------------
extern "C" void kernel_launch(void* const* d_in, const int* in_sizes, int n_in,
                              void* d_out, int out_size, void* d_ws, size_t ws_size,
                              hipStream_t stream) {
  (void)in_sizes; (void)n_in; (void)out_size; (void)ws_size;
  static const int   cinA[6]  = {0, 64, 128, 256, 512, 1024};
  static const int   midA[6]  = {45, 96, 192, 384, 768, 1536};
  static const int   coutA[6] = {64, 128, 256, 512, 1024, 2048};
  static const float rA[6]    = {0.1f, 0.2f, 0.2f, 0.4f, 0.4f, 0.4f};
  static const int   tkA[6]   = {1, 3, 3, 3, 3, 1};
  static const int   ssA[6]   = {2, 2, 1, 2, 1, 2};
  static const int   tsA[6]   = {1, 2, 1, 2, 1, 1};

  struct LP { const float *Wd, *Wf, *g, *b, *rm, *rv, *Wt, *bt; };
  int p = 0;
  const float* xyzs = (const float*)d_in[p++];
  LP lp[6];
  for (int i = 0; i < 6; ++i) {
    lp[i].Wd = (const float*)d_in[p++];
    lp[i].Wf = cinA[i] ? (const float*)d_in[p++] : nullptr;
    lp[i].g  = (const float*)d_in[p++];
    lp[i].b  = (const float*)d_in[p++];
    lp[i].rm = (const float*)d_in[p++];
    lp[i].rv = (const float*)d_in[p++];
    lp[i].Wt = (const float*)d_in[p++];
    lp[i].bt = (const float*)d_in[p++];
  }
  const float* fcW = (const float*)d_in[p++];
  const float* fcb = (const float*)d_in[p++];

  char* ws = (char*)d_ws;
  const size_t SLOT = (size_t)16 << 20;  // ping-pong persistent slots
  char* slot[2] = { ws, ws + SLOT };
  char* scratch0 = ws + 2 * SLOT;

  const float* xyz_cur = xyzs;
  const float* feats_cur = nullptr;
  int Lcur = 23, ncur = 2048;
  const float* feats_fin = nullptr;
  int lfin = 0, mfin = 0, cfin = 0;

  for (int i = 0; i < 6; ++i) {
    int cin = cinA[i], mid = midA[i], cout = coutA[i];
    int tk = tkA[i], ss = ssA[i], ts = tsA[i];
    int rad = tk / 2;
    int lout = (Lcur - 2 * rad - 1) / ts + 1;
    int m = ncur / ss;
    float r2 = rA[i] * rA[i];
    int N1 = 2 * lout * tk * m;   // always a multiple of 256 (m >= 128)
    int N2 = 2 * lout * m;        // always a multiple of 256
    int K1 = 3 + cin,  Kp1 = ((K1 + 31) / 32) * 32;
    int K2 = tk * mid, Kp2 = ((K2 + 31) / 32) * 32;
    int Mp1 = ((mid + 15) / 16) * 16;
    int Mp2 = ((cout + 15) / 16) * 16;

    // persistent outputs of this layer
    char* sp = slot[i & 1];
    float* anchor = (float*)sp;
    sp += ((size_t)2 * lout * m * 3 * 4 + 255) & ~(size_t)255;
    float* out2 = (float*)sp;  // feats [2][lout][m][cout]

    // scratch (bump; phase-2 reuses phase-1 region past out1)
    char* sc = scratch0;
    auto balloc = [&](size_t bytes) -> char* {
      char* q = sc; sc += (bytes + 255) & ~(size_t)255; return q;
    };
    float* out1 = (float*)balloc((size_t)N1 * mid * 4);
    char* reuse_mark = sc;
    int* fidx = (int*)balloc((size_t)2 * lout * m * 4);
    int* idx  = (int*)balloc((size_t)N1 * KNN * 4);
    unsigned short* B1t = (unsigned short*)balloc((size_t)N1 * Kp1 * 2);
    unsigned short* A1  = (unsigned short*)balloc((size_t)Mp1 * Kp1 * 2);

    fps_kernel<<<dim3(lout, 2), 256, 0, stream>>>(xyz_cur, Lcur, ncur, m, rad, ts, fidx, lout);
    int tot = 2 * lout * m;
    gather_anchor_k<<<(tot + 255) / 256, 256, 0, stream>>>(xyz_cur, fidx, anchor,
                                                           Lcur, ncur, m, rad, ts, lout);
    ball_query_k<<<dim3((m + 127) / 128, tk, 2 * lout), 128, 0, stream>>>(
        anchor, xyz_cur, idx, lout, tk, m, ncur, Lcur, ts, r2);
    pack_a1_k<<<(Mp1 * Kp1 + 255) / 256, 256, 0, stream>>>(lp[i].Wd, lp[i].Wf, A1,
                                                           mid, cin, Kp1, Mp1);
    pack_b1_k<<<dim3(N1), 128, 0, stream>>>(xyz_cur, feats_cur, anchor, idx, B1t,
                                            lout, tk, m, ncur, cin, Kp1, Lcur, ts);
    gemm_wmma_bf16<<<dim3(Mp1 / 16, N1 / 256), 128, 0, stream>>>(
        A1, B1t, out1, nullptr, mid, N1, Kp1, 0);

    sc = reuse_mark;  // phase-1 scratch dead after gemm1 (stream-ordered)
    unsigned short* B2t = (unsigned short*)balloc((size_t)N2 * Kp2 * 2);
    unsigned short* A2  = (unsigned short*)balloc((size_t)Mp2 * Kp2 * 2);
    pack_b2_k<<<dim3(N2), 128, 0, stream>>>(out1, lp[i].g, lp[i].b, lp[i].rm, lp[i].rv,
                                            B2t, lout, tk, m, mid, Kp2);
    pack_a2_k<<<(Mp2 * Kp2 + 255) / 256, 256, 0, stream>>>(lp[i].Wt, A2, cout, K2, Kp2, Mp2);
    gemm_wmma_bf16<<<dim3(Mp2 / 16, N2 / 256), 128, 0, stream>>>(
        A2, B2t, out2, lp[i].bt, cout, N2, Kp2, (i < 5) ? 1 : 0);

    xyz_cur = anchor; feats_cur = out2; Lcur = lout; ncur = m;
    feats_fin = out2; lfin = lout; mfin = m; cfin = cout;
  }

  // Head: mean over points, max over frames, FC (2x20 outputs, f32)
  float* fvec = (float*)scratch0;
  head_reduce_k<<<(2 * cfin + 255) / 256, 256, 0, stream>>>(feats_fin, fvec, lfin, mfin, cfin);
  head_fc_k<<<1, 64, 0, stream>>>(fvec, fcW, fcb, (float*)d_out, 20, cfin);
}
